// GQA_67010079752601
// MI455X (gfx1250) — matrix operand order repaired and checked by reading, not compiled, not run on hardware
//
#include <hip/hip_runtime.h>
#include <stdint.h>

typedef unsigned short u16;
typedef __attribute__((ext_vector_type(16))) __bf16        v16bf;
typedef __attribute__((ext_vector_type(8)))  float         v8f;
typedef __attribute__((ext_vector_type(4)))  unsigned int  v4u;
typedef __attribute__((ext_vector_type(4)))  float         v4f;

union FragU { v4u u[2]; v16bf v; };

#ifndef __has_builtin
#define __has_builtin(x) 0
#endif

#if __has_builtin(__builtin_amdgcn_global_load_async_to_lds_b128) && \
    __has_builtin(__builtin_amdgcn_s_wait_asynccnt)
#define HAS_ASYNC_LDS 1
#else
#define HAS_ASYNC_LDS 0
#endif

// Builtin signature (from hipcc diagnostic): param0 is
// 'int __attribute__((vector_size(16))) __device__ *'  -> AS1 int4*.
typedef int v4i_ __attribute__((vector_size(16)));
typedef __attribute__((address_space(1))) v4i_ GBuf128;   // global int4
typedef __attribute__((address_space(3))) v4i_ LBuf128;   // LDS int4

// 16B global -> LDS copy: async (ASYNCcnt) if available, else through VGPRs.
__device__ __forceinline__ void cp16_g2l(const u16* g, u16* l) {
#if HAS_ASYNC_LDS
  __builtin_amdgcn_global_load_async_to_lds_b128(
      (GBuf128*)(uintptr_t)g,                 // generic global == AS1 numerically
      (LBuf128*)(uint32_t)(uintptr_t)l,       // low 32 bits of generic = LDS offset
      0, 0);
#else
  *(v4u*)l = *(const v4u*)g;
#endif
}
__device__ __forceinline__ void wait_async_le8() {
#if HAS_ASYNC_LDS
  __builtin_amdgcn_s_wait_asynccnt(8);
#endif
}
__device__ __forceinline__ void wait_async_0() {
#if HAS_ASYNC_LDS
  __builtin_amdgcn_s_wait_asynccnt(0);
#endif
}

// f32 -> bf16 RNE, raw bits
__device__ __forceinline__ u16 f2bf(float f) {
  uint32_t u = __float_as_uint(f);
  return (u16)((u + 0x7FFFu + ((u >> 16) & 1u)) >> 16);
}
// packed pair: low 16 = a, high 16 = b
#if __has_builtin(__builtin_amdgcn_cvt_pk_bf16_f32)
typedef __attribute__((ext_vector_type(2))) __bf16 v2bf;
__device__ __forceinline__ uint32_t f2bf_pk(float a, float b) {
  union { v2bf v; uint32_t u; } c;
  c.v = __builtin_amdgcn_cvt_pk_bf16_f32(a, b);
  return c.u;
}
#else
__device__ __forceinline__ uint32_t f2bf_pk(float a, float b) {
  return (uint32_t)f2bf(a) | ((uint32_t)f2bf(b) << 16);
}
#endif

__device__ __forceinline__ v8f wmma_bf16(v16bf a, v16bf b, v8f c) {
  return __builtin_amdgcn_wmma_f32_16x16x32_bf16(false, a, false, b, (short)0, c,
                                                 false, false);
}

// A-fragment (16x32 bf16), row-major tile[m][k], row stride ld.
// Lane L: M=m+(L&15); halves 0-7 -> K=k0+(L>=16?8:0)+0..7, halves 8-15 -> +16.
__device__ __forceinline__ v16bf frag_a(const u16* t, int ld, int m, int k0, int lane) {
  const u16* base = t + (m + (lane & 15)) * ld + k0 + ((lane >> 4) << 3);
  FragU r;
  r.u[0] = *(const v4u*)(base);
  r.u[1] = *(const v4u*)(base + 16);
  return r.v;
}
// B-fragment (32x16 bf16) from K-contiguous layout tileT[n][k], row stride ld.
// Lane L: N=n+(L&15); halves 0-7 -> K=k0+(L>=16?8:0)+0..7, halves 8-15 -> +16.
__device__ __forceinline__ v16bf frag_b(const u16* t, int ld, int n, int k0, int lane) {
  const u16* base = t + (n + (lane & 15)) * ld + k0 + ((lane >> 4) << 3);
  FragU r;
  r.u[0] = *(const v4u*)(base);
  r.u[1] = *(const v4u*)(base + 16);
  return r.v;
}

// ---------------------------------------------------------------------------
// Tiled GEMM + bias: C = A @ W + bias, double-buffered LDS, K-step 32.
// Block tile 128x128, 256 thr = 8 waves (4M x 2N), each wave 32x64 = 8 WMMA/step.
// A_F32: A is f32 (cvt to bf16) else bf16 bits.
// OUT_MODE: 0 = f32 [row][col]; 1 = bf16 [row][col]; 2 = bf16 V-transposed
//           ([b][g][hd][token], for the attention V operand).
// ---------------------------------------------------------------------------
template <int A_F32, int OUT_MODE>
__global__ __launch_bounds__(256) void gemm_bias_kernel(
    const void* __restrict__ Ap, const float* __restrict__ W,
    const float* __restrict__ bias, void* __restrict__ Cp,
    int M, int N, int K, float out_scale) {
  __shared__ __align__(16) u16 As[2][128 * 40];   // [m][k], stride 40 halves
  __shared__ __align__(16) u16 WT[2][128 * 40];   // [n][k] transposed

  const int tid  = threadIdx.x;
  const int lane = tid & 31;
  const int wave = tid >> 5;
  const int m0 = blockIdx.y * 128;
  const int n0 = blockIdx.x * 128;
  const int wm = (wave & 3) * 32;
  const int wn = (wave >> 2) * 64;

  v4f a_f[4];
  v4u a_b[2];
  v4f w_f[4];

  auto g_load = [&](int k0) {
    if (A_F32) {
      const float* A = (const float*)Ap;
      int r = tid >> 3, seg = tid & 7;
#pragma unroll
      for (int it = 0; it < 4; ++it)
        a_f[it] = *(const v4f*)(A + (size_t)(m0 + r + it * 32) * K + k0 + seg * 4);
    } else {
      const u16* A = (const u16*)Ap;
      int r = tid >> 1, seg = tid & 1;
      const u16* p = A + (size_t)(m0 + r) * K + k0 + seg * 16;
      a_b[0] = *(const v4u*)(p);
      a_b[1] = *(const v4u*)(p + 8);
    }
    int kr = tid >> 3, seg = tid & 7;
    const float* wr = W + (size_t)(k0 + kr) * N + n0 + seg * 16;
#pragma unroll
    for (int j = 0; j < 4; ++j) w_f[j] = *(const v4f*)(wr + j * 4);
  };

  auto l_store = [&](int b) {
    if (A_F32) {
      int r = tid >> 3, seg = tid & 7;
#pragma unroll
      for (int it = 0; it < 4; ++it) {
        uint32_t* d = (uint32_t*)(As[b] + (r + it * 32) * 40 + seg * 4);
        d[0] = f2bf_pk(a_f[it].x, a_f[it].y);
        d[1] = f2bf_pk(a_f[it].z, a_f[it].w);
      }
    } else {
      int r = tid >> 1, seg = tid & 1;
      v4u* d = (v4u*)(As[b] + r * 40 + seg * 16);
      d[0] = a_b[0];
      d[1] = a_b[1];
    }
    int kr = tid >> 3, seg = tid & 7;
    u16* col = WT[b] + kr;
    int nb = seg * 16;
#pragma unroll
    for (int j = 0; j < 4; ++j) {
      uint32_t p0 = f2bf_pk(w_f[j].x, w_f[j].y);
      uint32_t p1 = f2bf_pk(w_f[j].z, w_f[j].w);
      col[(nb + j * 4 + 0) * 40] = (u16)p0;
      col[(nb + j * 4 + 1) * 40] = (u16)(p0 >> 16);
      col[(nb + j * 4 + 2) * 40] = (u16)p1;
      col[(nb + j * 4 + 3) * 40] = (u16)(p1 >> 16);
    }
  };

  v8f acc[2][4];
#pragma unroll
  for (int i = 0; i < 2; ++i)
#pragma unroll
    for (int j = 0; j < 4; ++j) acc[i][j] = {};

  g_load(0);
  l_store(0);
  __syncthreads();

  const int nk = K >> 5;
  for (int ki = 0; ki < nk; ++ki) {
    const int cur = ki & 1;
    if (ki + 1 < nk) g_load((ki + 1) << 5);   // overlap with WMMAs below

    v16bf a0 = frag_a(As[cur], 40, wm, 0, lane);
    v16bf a1 = frag_a(As[cur], 40, wm + 16, 0, lane);
#pragma unroll
    for (int j = 0; j < 4; ++j) {
      v16bf b = frag_b(WT[cur], 40, wn + j * 16, 0, lane);
      acc[0][j] = wmma_bf16(a0, b, acc[0][j]);
      acc[1][j] = wmma_bf16(a1, b, acc[1][j]);
    }
    if (ki + 1 < nk) l_store(1 - cur);
    __syncthreads();
  }

  // epilogue; C layout: M = v + 8*(lane/16), N = lane%16
  float bcol[4];
#pragma unroll
  for (int j = 0; j < 4; ++j) bcol[j] = bias[n0 + wn + j * 16 + (lane & 15)];
#pragma unroll
  for (int i = 0; i < 2; ++i)
#pragma unroll
    for (int j = 0; j < 4; ++j)
#pragma unroll
      for (int v = 0; v < 8; ++v) {
        int row = m0 + wm + i * 16 + v + ((lane >> 4) << 3);
        int col = n0 + wn + j * 16 + (lane & 15);
        float val = (acc[i][j][v] + bcol[j]) * out_scale;
        if (OUT_MODE == 0) {
          ((float*)Cp)[(size_t)row * N + col] = val;
        } else if (OUT_MODE == 1) {
          ((u16*)Cp)[(size_t)row * N + col] = f2bf(val);
        } else {  // V transposed: [b][g][hd][token], N == 512
          int b = row >> 11, s = row & 2047;
          int g = col >> 7, hd = col & 127;
          ((u16*)Cp)[(((size_t)(b * 4 + g) * 128 + hd) << 11) + s] = f2bf(val);
        }
      }
}

// ---------------------------------------------------------------------------
// Flash-attention GQA. Grid (qtile=16, head=16, batch=2), 256 thr.
// Double-buffered K/V^T tiles via async global->LDS; Q frags in registers;
// online softmax; P round-trips per-wave LDS for C->A layout.
// ---------------------------------------------------------------------------
__global__ __launch_bounds__(256) void gqa_attn_kernel(
    const u16* __restrict__ Qb, const u16* __restrict__ Kb,
    const u16* __restrict__ Vt, u16* __restrict__ Ob) {
  const int S = 2048, HD = 128, DQ = 2048, DKV = 512;
  __shared__ __align__(16) u16 Kt[2][64 * 136];    // [key][hd]
  __shared__ __align__(16) u16 VT[2][128 * 72];    // [hd][key]
  __shared__ __align__(16) u16 Pt[8 * 16 * 72];    // per-wave [q][key]

  const int tid = threadIdx.x, lane = tid & 31, wave = tid >> 5;
  const int qt = blockIdx.x, head = blockIdx.y, bb = blockIdx.z;
  const int g = head >> 2;  // R = 4 q-heads per kv group

  const u16* Qhead = Qb + (size_t)bb * S * DQ + head * HD;
  const u16* Kkv   = Kb + (size_t)bb * S * DKV + g * HD;
  const u16* Vth   = Vt + ((size_t)(bb * 4 + g) * HD) * S;  // [hd][token]
  const int qbase = qt * 128 + wave * 16;

  auto issue_tile = [&](int kt, int buf) {  // 8 x 16B per thread
    {
      int row = tid >> 2, seg = tid & 3;
      const u16* gp = Kkv + (size_t)(kt * 64 + row) * DKV + seg * 32;
      u16* lp = Kt[buf] + row * 136 + seg * 32;
#pragma unroll
      for (int i = 0; i < 4; ++i) cp16_g2l(gp + i * 8, lp + i * 8);
    }
    {
      int row = tid >> 1, seg = tid & 1;
      const u16* gp = Vth + (size_t)row * S + kt * 64 + seg * 32;
      u16* lp = VT[buf] + row * 72 + seg * 32;
#pragma unroll
      for (int i = 0; i < 4; ++i) cp16_g2l(gp + i * 8, lp + i * 8);
    }
  };

  v16bf qf[4];
#pragma unroll
  for (int ks = 0; ks < 4; ++ks) qf[ks] = frag_a(Qhead, DQ, qbase, ks * 32, lane);

  v8f o[8];
#pragma unroll
  for (int i = 0; i < 8; ++i) o[i] = {};
  float mrow[8], lrow[8];
#pragma unroll
  for (int v = 0; v < 8; ++v) { mrow[v] = -3.0e38f; lrow[v] = 0.f; }

  u16* Pw = Pt + wave * 16 * 72;

  issue_tile(0, 0);
  const int NT = S / 64;
  for (int kt = 0; kt < NT; ++kt) {
    const int cur = kt & 1;
    if (kt + 1 < NT) {
      issue_tile(kt + 1, 1 - cur);  // prefetch next tile
      wait_async_le8();             // in-order: all but newest 8 complete
    } else {
      wait_async_0();
    }
    __syncthreads();                // tile kt visible to all waves

    // S = Q @ K^T (scale pre-folded into Q projection)
    v8f s[4];
#pragma unroll
    for (int nt = 0; nt < 4; ++nt) {
      v8f a = {};
#pragma unroll
      for (int ks = 0; ks < 4; ++ks) {
        v16bf kb = frag_b(Kt[cur], 136, nt * 16, ks * 32, lane);
        a = wmma_bf16(qf[ks], kb, a);
      }
      s[nt] = a;
    }

    // online softmax; row M = v + 8*(lane/16), reduce across 16 lanes
#pragma unroll
    for (int v = 0; v < 8; ++v) {
      float mx = fmaxf(fmaxf(s[0][v], s[1][v]), fmaxf(s[2][v], s[3][v]));
#pragma unroll
      for (int d = 1; d < 16; d <<= 1) mx = fmaxf(mx, __shfl_xor(mx, d, 32));
      float mnew = fmaxf(mrow[v], mx);
      float corr = __expf(mrow[v] - mnew);
      float rs = 0.f;
#pragma unroll
      for (int nt = 0; nt < 4; ++nt) {
        float p = __expf(s[nt][v] - mnew);
        s[nt][v] = p;
        rs += p;
      }
#pragma unroll
      for (int d = 1; d < 16; d <<= 1) rs += __shfl_xor(rs, d, 32);
      lrow[v] = lrow[v] * corr + rs;
      mrow[v] = mnew;
#pragma unroll
      for (int ot = 0; ot < 8; ++ot) o[ot][v] *= corr;
    }

    // P regs (C layout) -> per-wave LDS tile; then O += P @ V
    {
      int pr = (lane >> 4) << 3;
#pragma unroll
      for (int v = 0; v < 8; ++v)
#pragma unroll
        for (int nt = 0; nt < 4; ++nt)
          Pw[(v + pr) * 72 + nt * 16 + (lane & 15)] = f2bf(s[nt][v]);
    }
#pragma unroll
    for (int kc = 0; kc < 2; ++kc) {
      v16bf pa = frag_a(Pw, 72, 0, kc * 32, lane);
#pragma unroll
      for (int ot = 0; ot < 8; ++ot) {
        v16bf vb = frag_b(VT[cur], 72, ot * 16, kc * 32, lane);
        o[ot] = wmma_bf16(pa, vb, o[ot]);
      }
    }
    __syncthreads();  // all waves done reading buf cur before it is rewritten
  }

  // finalize + store bf16 [token][head*128 + hd]
#pragma unroll
  for (int v = 0; v < 8; ++v) lrow[v] = 1.f / lrow[v];
  const size_t obase = (size_t)bb * S * DQ;
#pragma unroll
  for (int ot = 0; ot < 8; ++ot)
#pragma unroll
    for (int v = 0; v < 8; ++v) {
      int row = qbase + v + ((lane >> 4) << 3);
      int col = head * HD + ot * 16 + (lane & 15);
      Ob[obase + (size_t)row * DQ + col] = f2bf(o[ot][v] * lrow[v]);
    }
}

// ---------------------------------------------------------------------------
extern "C" void kernel_launch(void* const* d_in, const int* in_sizes, int n_in,
                              void* d_out, int out_size, void* d_ws, size_t ws_size,
                              hipStream_t stream) {
  (void)in_sizes; (void)n_in; (void)out_size; (void)ws_size;
  const float* x  = (const float*)d_in[0];
  const float* Wq = (const float*)d_in[1];
  const float* bq = (const float*)d_in[2];
  const float* Wk = (const float*)d_in[3];
  const float* bk = (const float*)d_in[4];
  const float* Wv = (const float*)d_in[5];
  const float* bv = (const float*)d_in[6];
  const float* Wo = (const float*)d_in[7];
  const float* bo = (const float*)d_in[8];
  float* out = (float*)d_out;

  const int M = 4096;   // B*S tokens
  const int D = 2048;   // model dim
  const int NKV = 512;  // G*HD
  const float qscale = 0.08838834764831845f;  // 1/sqrt(128), folded into Q proj

  u16* Qb = (u16*)d_ws;                 // [4096][2048] bf16
  u16* Kb = Qb + (size_t)M * D;         // [4096][512]
  u16* Vt = Kb + (size_t)M * NKV;       // [B][G][128][2048] (V transposed)
  u16* Ob = Vt + (size_t)M * NKV;       // [4096][2048] attention output

  dim3 blk(256);
  gemm_bias_kernel<1, 1><<<dim3(D / 128, M / 128), blk, 0, stream>>>(x, Wq, bq, Qb, M, D, D, qscale);
  gemm_bias_kernel<1, 1><<<dim3(NKV / 128, M / 128), blk, 0, stream>>>(x, Wk, bk, Kb, M, NKV, D, 1.f);
  gemm_bias_kernel<1, 2><<<dim3(NKV / 128, M / 128), blk, 0, stream>>>(x, Wv, bv, Vt, M, NKV, D, 1.f);
  gqa_attn_kernel<<<dim3(16, 16, 2), blk, 0, stream>>>(Qb, Kb, Vt, Ob);
  gemm_bias_kernel<0, 0><<<dim3(D / 128, M / 128), blk, 0, stream>>>(Ob, Wo, bo, out, M, D, D, 1.f);
}
